// RouterBlock_78460462563549
// MI455X (gfx1250) — matrix operand order repaired and checked by reading, not compiled, not run on hardware
//
#include <hip/hip_runtime.h>
#include <hip/hip_bf16.h>

#define Bz 8
#define Sz 4096
#define Dz 768
#define Ez 8
#define ITz 12
#define Fz (Dz + ITz)      // 780
#define BSz (Bz * Sz)      // 32768

#define BN 64                          // N-tile per block
#define SLAB_ELEMS (BN * Dz)           // 49152 bf16 elements = 98304 bytes
#define SLAB_BYTES (SLAB_ELEMS * 2)

typedef __attribute__((ext_vector_type(16))) __bf16 v16bf;
typedef __attribute__((ext_vector_type(8)))  __bf16 v8bf;
typedef __attribute__((ext_vector_type(8)))  float  v8f;
typedef __attribute__((ext_vector_type(4)))  unsigned int u32x4;
typedef __attribute__((ext_vector_type(8)))  int      i32x8;
typedef __attribute__((ext_vector_type(4)))  int      i32x4;

#if defined(__has_builtin)
#if __has_builtin(__builtin_amdgcn_tensor_load_to_lds)
#define HAVE_TDM 1
#endif
#endif
#if defined(__has_include)
#if __has_include(<hip/amd_detail/amd_gfx1250_TDM.h>)
#define TDM_6ARG 1                      // therock-10.0 headers => 6-arg builtin
#endif
#endif

__device__ __forceinline__ unsigned short f32_to_bf16_rne(float f) {
    union { float f; unsigned u; } cv; cv.f = f;
    unsigned u = cv.u;
    unsigned r = (u + 0x7FFFu + ((u >> 16) & 1u)) >> 16;
    return (unsigned short)r;
}

// ---------------------------------------------------------------------------
// Kernel 1: mask-mult + LayerNorm(780) + router linear + softmax, fused with
// f32->bf16 conversion of masked hs (GEMM A matrix). One wave32 per token.
// ---------------------------------------------------------------------------
__global__ __launch_bounds__(256) void router_kernel(
    const float* __restrict__ hs, const int* __restrict__ am,
    const int* __restrict__ iter_p, const float* __restrict__ gamma,
    const float* __restrict__ beta, const float* __restrict__ rw,
    unsigned short* __restrict__ hs_bf16, float* __restrict__ probs_out)
{
    const int wave = threadIdx.x >> 5;
    const int lane = threadIdx.x & 31;
    const int t = blockIdx.x * 8 + wave;
    if (t >= BSz) return;

    const float mask = (float)am[t];
    const int   iteration = *iter_p;
    const float* __restrict__ row = hs + (size_t)t * Dz;

    float vals[25];
    float sum = 0.f, sumsq = 0.f;
#pragma unroll
    for (int j = 0; j < 25; ++j) {
        const int i = lane + 32 * j;
        float v = 0.f;
        if (i < Dz) {
            v = row[i] * mask;
            hs_bf16[(size_t)t * Dz + i] = f32_to_bf16_rne(v);
        } else if (i < Fz) {
            v = (i - Dz == iteration) ? 1.f : 0.f;
        }
        vals[j] = v;
        sum   += v;
        sumsq += v * v;
    }
#pragma unroll
    for (int o = 16; o > 0; o >>= 1) {
        sum   += __shfl_xor(sum,   o, 32);
        sumsq += __shfl_xor(sumsq, o, 32);
    }
    const float rF   = 1.0f / (float)Fz;
    const float mu   = sum * rF;
    const float var  = sumsq * rF - mu * mu;
    const float rstd = rsqrtf(var + 1e-5f);

    float lg[Ez];
#pragma unroll
    for (int e = 0; e < Ez; ++e) lg[e] = 0.f;
#pragma unroll
    for (int j = 0; j < 25; ++j) {
        const int i = lane + 32 * j;
        if (i < Fz) {
            const float x = (vals[j] - mu) * rstd * gamma[i] + beta[i];
#pragma unroll
            for (int e = 0; e < Ez; ++e) lg[e] += x * rw[e * Fz + i];
        }
    }
#pragma unroll
    for (int e = 0; e < Ez; ++e) {
#pragma unroll
        for (int o = 16; o > 0; o >>= 1) lg[e] += __shfl_xor(lg[e], o, 32);
    }
    float mx = lg[0];
#pragma unroll
    for (int e = 1; e < Ez; ++e) mx = fmaxf(mx, lg[e]);
    float p[Ez]; float se = 0.f;
#pragma unroll
    for (int e = 0; e < Ez; ++e) { p[e] = __expf(lg[e] - mx); se += p[e]; }
    const float rs = 1.0f / se;
    if (lane < Ez) probs_out[(size_t)t * Ez + lane] = p[lane] * rs;
}

// ---------------------------------------------------------------------------
// Kernel 2: expert_w [E,K,N] f32 -> transposed bf16 [E,N,K] so that B-matrix
// fragments are contiguous and a whole N-slab is one contiguous 98KB block
// (perfect for a 1-D TDM descriptor).
// ---------------------------------------------------------------------------
__global__ __launch_bounds__(256) void wconvert_kernel(
    const float* __restrict__ W, unsigned short* __restrict__ Wt)
{
    const size_t idx = (size_t)blockIdx.x * 256 + threadIdx.x; // E*D*D total
    const int e = (int)(idx / (Dz * Dz));
    const int r = (int)(idx % (Dz * Dz));
    const int n = r / Dz;
    const int k = r % Dz;
    Wt[idx] = f32_to_bf16_rne(W[((size_t)e * Dz + k) * Dz + n]);
}

// ---------------------------------------------------------------------------
// TDM: DMA one contiguous 98KB B-slab (bf16, [64 cols][768 K]) global -> LDS.
// D# per CDNA5 ISA ch.8: group0 = {count=1 | lds_addr | global_addr | type=2},
// group1 = {data_size=2B, tensor_dim0 = tile_dim0 = 49152, tensor_dim1 = 1}.
// workgroup_mask = 0 (not in a cluster). Tracked by TENSORcnt.
// ---------------------------------------------------------------------------
#if HAVE_TDM
__device__ __forceinline__ void tdm_load_slab(const unsigned short* gsrc,
                                              unsigned lds_byte_off)
{
    const unsigned long long ga = (unsigned long long)(size_t)gsrc;
    u32x4 g0;
    g0[0] = 1u;                                     // count=1 (valid), user mode
    g0[1] = lds_byte_off;                           // lds_addr
    g0[2] = (unsigned)(ga & 0xFFFFFFFFu);           // global_addr[31:0]
    g0[3] = (unsigned)((ga >> 32) & 0x01FFFFFFu)    // global_addr[56:32]
            | 0x80000000u;                          // type=2 ("image")
    i32x8 g1;
    g1[0] = 0x00010000;                             // data_size=1 (2 bytes)
    g1[1] = (int)0xC0000000;                        // tensor_dim0[15:0]=49152 <<16
    g1[2] = 0x00010000;                             // tensor_dim0[31:16]=0 | tensor_dim1=1
    g1[3] = (int)0xC0000000;                        // tile_dim0=49152 <<16
    g1[4] = 0;                                      // tile_dim1=0, tile_dim2=0 (unused)
    g1[5] = (int)SLAB_ELEMS;                        // tensor_dim0_stride[31:0]
    g1[6] = 0;
    g1[7] = 0;
    const i32x4 z4 = {0, 0, 0, 0};
#if TDM_6ARG
    const i32x8 z8 = {0, 0, 0, 0, 0, 0, 0, 0};
    __builtin_amdgcn_tensor_load_to_lds(g0, g1, z4, z4, z8, 0);
#else
    __builtin_amdgcn_tensor_load_to_lds(g0, g1, z4, z4, 0);
#endif
}
#endif

// ---------------------------------------------------------------------------
// Kernel 3: out[m,n] = sum_e probs[m,e] * (A[m,:] @ W_e[:,n] + b_e[n])
// 8 waves/block, 128x64 tile/block. B slabs double-buffered in LDS via TDM
// (next expert's DMA overlapped with current expert's WMMAs); A fragments
// streamed from global with prefetch. bf16 WMMA 16x16x32, f32 accumulate.
// ---------------------------------------------------------------------------
__global__ __launch_bounds__(256) void moe_gemm_kernel(
    const unsigned short* __restrict__ Abf, const unsigned short* __restrict__ Wtb,
    const float* __restrict__ eb, const float* __restrict__ probs,
    float* __restrict__ out)
{
    extern __shared__ unsigned short smem[];        // 2 x SLAB_ELEMS bf16

    const int wave = threadIdx.x >> 5;
    const int lane = threadIdx.x & 31;
    const int half = lane >> 4;     // 0: lanes 0-15, 1: lanes 16-31
    const int lidx = lane & 15;
    const int m0 = blockIdx.x * 128 + wave * 16;
    const int n0 = blockIdx.y * BN;

    const __bf16* __restrict__ A = (const __bf16*)Abf;
    const unsigned lds_base = (unsigned)(unsigned long long)(size_t)&smem[0];

    const v8f zero8 = {0.f, 0.f, 0.f, 0.f, 0.f, 0.f, 0.f, 0.f};
    v8f fin[4];
#pragma unroll
    for (int sn = 0; sn < 4; ++sn) fin[sn] = zero8;

    // A fragment (16-bit A 16x32): lane<16 row=lidx holds K k..k+7 & k+16..k+23,
    // lane>=16 holds K k+8..k+15 & k+24..k+31  -> base offset half*8, +16.
    const __bf16* __restrict__ arow = A + (size_t)(m0 + lidx) * Dz + half * 8;

#if HAVE_TDM
    if (wave == 0) tdm_load_slab(Wtb + (size_t)n0 * Dz, lds_base);
#endif

    for (int e = 0; e < Ez; ++e) {
        const int buf = e & 1;
#if HAVE_TDM
        if (wave == 0) __builtin_amdgcn_s_wait_tensorcnt(0);  // buf[e&1] ready
        __syncthreads();   // all waves see slab; prev buffer no longer being read
        if (wave == 0 && e + 1 < Ez)
            tdm_load_slab(Wtb + ((size_t)(e + 1) * Dz + n0) * Dz,
                          lds_base + (unsigned)((buf ^ 1) * SLAB_BYTES));
#else
        // Fallback: cooperative global->LDS copy of the 98KB slab.
        __syncthreads();
        {
            const unsigned short* gsrc = Wtb + ((size_t)e * Dz + n0) * Dz;
            for (int i = threadIdx.x; i < SLAB_ELEMS / 8; i += 256)
                *(v8bf*)&smem[buf * SLAB_ELEMS + i * 8] =
                    *(const v8bf*)&gsrc[i * 8];
        }
        __syncthreads();
#endif
        const unsigned short* __restrict__ bslab = &smem[buf * SLAB_ELEMS];

        v8f acc[4];
#pragma unroll
        for (int sn = 0; sn < 4; ++sn) acc[sn] = zero8;

        for (int k = 0; k < Dz; k += 32) {
            __builtin_prefetch(arow + k + 32, 0, 1);
            const v8bf alo = *(const v8bf*)(arow + k);
            const v8bf ahi = *(const v8bf*)(arow + k + 16);
            const v16bf a = __builtin_shufflevector(alo, ahi,
                0, 1, 2, 3, 4, 5, 6, 7, 8, 9, 10, 11, 12, 13, 14, 15);
#pragma unroll
            for (int sn = 0; sn < 4; ++sn) {
                // B fragment (16-bit B 32x16): lane<16 col=lidx K k..k+7 & k+16..k+23,
                // lane>=16 K k+8..k+15 & k+24..k+31; matches A fragment order.
                const unsigned short* bp =
                    bslab + (sn * 16 + lidx) * Dz + half * 8 + k;
                const v8bf blo = *(const v8bf*)(bp);
                const v8bf bhi = *(const v8bf*)(bp + 16);
                const v16bf b = __builtin_shufflevector(blo, bhi,
                    0, 1, 2, 3, 4, 5, 6, 7, 8, 9, 10, 11, 12, 13, 14, 15);
                acc[sn] = __builtin_amdgcn_wmma_f32_16x16x32_bf16(
                    false, a, false, b, (short)0, acc[sn], false, false);
            }
        }
        // C/D layout: VGPR i holds (M = i + 8*half, N = lidx + sn*16 + n0)
#pragma unroll
        for (int i = 0; i < 8; ++i) {
            const float pe = probs[(size_t)(m0 + i + 8 * half) * Ez + e];
#pragma unroll
            for (int sn = 0; sn < 4; ++sn) {
                const float be = eb[e * Dz + n0 + sn * 16 + lidx];
                fin[sn][i] += pe * (acc[sn][i] + be);
            }
        }
    }
#pragma unroll
    for (int sn = 0; sn < 4; ++sn) {
        const int n = n0 + sn * 16 + lidx;
#pragma unroll
        for (int i = 0; i < 8; ++i) {
            out[(size_t)(m0 + i + 8 * half) * Dz + n] = fin[sn][i];
        }
    }
}

// ---------------------------------------------------------------------------
extern "C" void kernel_launch(void* const* d_in, const int* in_sizes, int n_in,
                              void* d_out, int out_size, void* d_ws, size_t ws_size,
                              hipStream_t stream) {
    const float* hs    = (const float*)d_in[0];
    const int*   am    = (const int*)d_in[1];
    const int*   itp   = (const int*)d_in[2];
    const float* gamma = (const float*)d_in[3];
    const float* beta  = (const float*)d_in[4];
    const float* rw    = (const float*)d_in[5];
    const float* ew    = (const float*)d_in[6];
    const float* eb    = (const float*)d_in[7];

    float* out   = (float*)d_out;
    float* probs = out + (size_t)BSz * Dz;            // second tuple element

    // workspace: hs as bf16 (50.3 MB) + transposed W as bf16 (9.4 MB)
    unsigned short* hsb = (unsigned short*)d_ws;
    unsigned short* wtb = hsb + (size_t)BSz * Dz;

    // 2 x 98KB LDS double buffer (WGP has 320KB)
    (void)hipFuncSetAttribute((const void*)moe_gemm_kernel,
                              hipFuncAttributeMaxDynamicSharedMemorySize,
                              2 * SLAB_BYTES);

    router_kernel<<<BSz / 8, 256, 0, stream>>>(hs, am, itp, gamma, beta, rw, hsb, probs);
    wconvert_kernel<<<(Ez * Dz * Dz) / 256, 256, 0, stream>>>(ew, wtb);
    moe_gemm_kernel<<<dim3(BSz / 128, Dz / BN), 256, 2 * SLAB_BYTES, stream>>>(
        hsb, wtb, eb, probs, out);
}
